// WeightOnlyInt8Linear_38139309588766
// MI455X (gfx1250) — matrix operand order repaired and checked by reading, not compiled, not run on hardware
//
#include <hip/hip_runtime.h>
#include <hip/hip_bf16.h>

typedef __attribute__((ext_vector_type(16))) __bf16 v16bf;
typedef __attribute__((ext_vector_type(8)))  __bf16 v8bf;
typedef __attribute__((ext_vector_type(8)))  float  v8f;
typedef __attribute__((ext_vector_type(4)))  int    v4i;   // native clang vector (NT-load capable)

#define IN_F  4096
#define OUT_F 32000
#define BATCH 32

// One wave per 16-wide N tile, both M tiles (M=32) per wave so each converted
// int8->bf16 weight fragment feeds two v_wmma_f32_16x16x32_bf16 ops.
// Weight stream (524 MB of int32) is the roofline limiter (~22.5 us at
// 23.3 TB/s) -> 4x b128 NT loads per fragment, fully coalesced along K,
// bypassing L2 retention so the 256 KB A matrix stays cache-resident.
__global__ __launch_bounds__(256)
void wo_int8_linear_kernel(const __bf16* __restrict__ X,   // [32, 4096] bf16
                           const int*    __restrict__ W,   // [32000, 4096] int32 (int8-valued)
                           const __bf16* __restrict__ S,   // [32000] bf16
                           __bf16*       __restrict__ O)   // [32, 32000] bf16
{
    const int lane  = threadIdx.x & 31;
    const int wave  = threadIdx.x >> 5;
    const int nTile = blockIdx.x * 8 + wave;     // 2000 tiles of 16 columns
    const int n0    = nTile * 16;

    const int nl = lane & 15;                    // local column / local row id
    const int hi = lane >> 4;                    // half-wave select

    const int n = n0 + nl;

    // A fragment bases: row m = nl (+16 for second M tile), K chunk per half-wave
    const __bf16* xr0 = X + (size_t)nl        * IN_F + hi * 8;
    const __bf16* xr1 = X + (size_t)(nl + 16) * IN_F + hi * 8;
    // B fragment base: W row n (column of B), K chunks 8*hi..+7 and 16+8*hi..+7 per lane
    const int* wr = W + (size_t)n * IN_F + hi * 8;

    v8f acc0 = {};   // M rows 0..15
    v8f acc1 = {};   // M rows 16..31

    #pragma unroll 4
    for (int k0 = 0; k0 < IN_F; k0 += 32) {
        // ---- B fragment: two runs of 8 consecutive int32 -> bf16 (exact for |w|<=127).
        // Non-temporal: single-use 524 MB stream, do not retain in L2.
        v4i w0 = __builtin_nontemporal_load((const v4i*)(wr + k0 +  0));
        v4i w1 = __builtin_nontemporal_load((const v4i*)(wr + k0 +  4));
        v4i w2 = __builtin_nontemporal_load((const v4i*)(wr + k0 + 16));
        v4i w3 = __builtin_nontemporal_load((const v4i*)(wr + k0 + 20));
        v16bf b;
        b[ 0] = (__bf16)(float)w0.x; b[ 1] = (__bf16)(float)w0.y;
        b[ 2] = (__bf16)(float)w0.z; b[ 3] = (__bf16)(float)w0.w;
        b[ 4] = (__bf16)(float)w1.x; b[ 5] = (__bf16)(float)w1.y;
        b[ 6] = (__bf16)(float)w1.z; b[ 7] = (__bf16)(float)w1.w;
        b[ 8] = (__bf16)(float)w2.x; b[ 9] = (__bf16)(float)w2.y;
        b[10] = (__bf16)(float)w2.z; b[11] = (__bf16)(float)w2.w;
        b[12] = (__bf16)(float)w3.x; b[13] = (__bf16)(float)w3.y;
        b[14] = (__bf16)(float)w3.z; b[15] = (__bf16)(float)w3.w;

        // ---- A fragments: two 16-byte chunks per lane per M tile (L2 hits)
        v8bf a0lo = *(const v8bf*)(xr0 + k0);
        v8bf a0hi = *(const v8bf*)(xr0 + k0 + 16);
        v8bf a1lo = *(const v8bf*)(xr1 + k0);
        v8bf a1hi = *(const v8bf*)(xr1 + k0 + 16);
        v16bf a0 = __builtin_shufflevector(a0lo, a0hi,
                     0,1,2,3,4,5,6,7, 8,9,10,11,12,13,14,15);
        v16bf a1 = __builtin_shufflevector(a1lo, a1hi,
                     0,1,2,3,4,5,6,7, 8,9,10,11,12,13,14,15);

        // D = A x B + C   (8 args: neg_a, A, neg_b, B, c_mod, C, reuse_a, reuse_b)
        acc0 = __builtin_amdgcn_wmma_f32_16x16x32_bf16(false, a0, false, b,
                                                       (short)0, acc0, false, false);
        acc1 = __builtin_amdgcn_wmma_f32_16x16x32_bf16(false, a1, false, b,
                                                       (short)0, acc1, false, false);
    }

    // Epilogue: per-output-channel scale, f32 -> bf16, strided NT stores
    // (2 MB written once, never re-read by this kernel).
    const float s = (float)S[n];
    const int mbase = hi * 8;
    #pragma unroll
    for (int r = 0; r < 8; ++r) {
        __bf16 o0 = (__bf16)(acc0[r] * s);
        __bf16 o1 = (__bf16)(acc1[r] * s);
        __builtin_nontemporal_store(o0, O + (size_t)(mbase + r)      * OUT_F + n);
        __builtin_nontemporal_store(o1, O + (size_t)(mbase + r + 16) * OUT_F + n);
    }
}

extern "C" void kernel_launch(void* const* d_in, const int* in_sizes, int n_in,
                              void* d_out, int out_size, void* d_ws, size_t ws_size,
                              hipStream_t stream) {
    const __bf16* X = (const __bf16*)d_in[0];
    const int*    W = (const int*)d_in[1];
    const __bf16* S = (const __bf16*)d_in[2];
    __bf16*       O = (__bf16*)d_out;

    // 8 waves/block * 16 cols/wave = 128 cols/block; 32000/128 = 250 blocks.
    dim3 grid(OUT_F / 128);
    dim3 block(256);
    wo_int8_linear_kernel<<<grid, block, 0, stream>>>(X, W, S, O);
}
